// LowrankLearnableHash_55886114456011
// MI455X (gfx1250) — matrix operand order repaired and checked by reading, not compiled, not run on hardware
//
#include <hip/hip_runtime.h>
#include <hip/hip_bf16.h>
#include <math.h>

#define NRAYS   8192
#define NSAMP   256
#define NPTS    (NRAYS * NSAMP)
#define FEATD   32
#define RPLANE  128
#define RFEAT   128
#define NFRAG   16   // pre-baked weight B-fragments

typedef __attribute__((ext_vector_type(16))) _Float16 v16h;
typedef __attribute__((ext_vector_type(8)))  _Float16 v8h;
typedef __attribute__((ext_vector_type(8)))  float    v8f;

__constant__ constexpr float kRadius = 1.3f;
__constant__ constexpr float kDT = 2.0f * 1.3f * 1.7320508075688772f / 256.0f;

// ---------------- WMMA helpers (ISA 7.12.2 layouts, wave32) -----------------
// A-fragment element map (16-bit A 16x32): element e of lane (m = lane&15,
// hi = lane>>4) holds K = (e>>3)*16 + hi*8 + (e&7), row M = m.
// B-fragment: lane holds column n = lane&15; element e holds K = (e>>3)*16 + hi*8 + (e&7).
// C/D f32: VGPR v holds D[v + 8*hi][lane&15].

__device__ __forceinline__ v8f wmma_f16(v16h a, v16h b, v8f c) {
  return __builtin_amdgcn_wmma_f32_16x16x32_f16(
      false, a, false, b, (short)0, c, false, false);
}

// Build a B fragment from fp32 weights W[k][col] (ld = cols). Used in prep only.
__device__ __forceinline__ v16h make_b16(const float* W, int ld, int kstart,
                                         int kvalid, int col, int ncols) {
  v16h b;
#pragma unroll
  for (int e = 0; e < 16; ++e) {
    int k = (kstart & ~31) + ((kstart >> 4) & 1) * 8 + ((e >> 3) << 4) + (e & 7);
    float val = (k < kvalid && col < ncols) ? W[k * ld + col] : 0.0f;
    b[e] = (_Float16)val;
  }
  return b;
}

// ---------------- weight fragment prep (once per launch) --------------------
// Fragment table (all per-lane, f16, exact WMMA B layout):
//  0..3 : Ws1  [32x64]  N-tile nt            (K 0..31)
//  4,5  : Ws2  [64x1]   K-lo / K-hi          (N padded to 16)
//  6..9 : Wc1  [35x64]  N-tile nt, K-chunk 0 (rows 0..31)
// 10..13: Wc1  [35x64]  N-tile nt, K-chunk 1 (rows 32..34 + zero pad)
// 14,15 : Wc2  [64x3]   K-lo / K-hi          (N padded to 16)
__global__ __launch_bounds__(512) void prep_weights(
    const float* __restrict__ Ws1, const float* __restrict__ Ws2,
    const float* __restrict__ Wc1, const float* __restrict__ Wc2,
    _Float16* __restrict__ wfrag) {
  int t = threadIdx.x;             // 512 = NFRAG * 32
  int frag = t >> 5, lane = t & 31;
  int n = lane & 15, hi = lane >> 4;
  v16h b;
  if (frag < 4) {
    b = make_b16(Ws1, 64, hi * 16, 32, frag * 16 + n, 64);
  } else if (frag == 4) {
    b = make_b16(Ws2, 1, hi * 16, 64, n, 1);
  } else if (frag == 5) {
    b = make_b16(Ws2, 1, 32 + hi * 16, 64, n, 1);
  } else if (frag < 10) {
    b = make_b16(Wc1, 64, hi * 16, 35, (frag - 6) * 16 + n, 64);
  } else if (frag < 14) {
    b = make_b16(Wc1, 64, 32 + hi * 16, 35, (frag - 10) * 16 + n, 64);
  } else if (frag == 14) {
    b = make_b16(Wc2, 3, hi * 16, 64, n, 3);
  } else {
    b = make_b16(Wc2, 3, 32 + hi * 16, 64, n, 3);
  }
  *(v16h*)(wfrag + (size_t)(frag * 32 + lane) * 16) = b;
}

__device__ __forceinline__ v16h load_frag(const _Float16* __restrict__ wf,
                                          int frag, int lane) {
  return *(const v16h*)(wf + (size_t)(frag * 32 + lane) * 16);
}

// ---------------- interpolation helper --------------------------------------
__device__ __forceinline__ void bilerp_setup(float cx, float cy,
                                             int& ix0, int& ix1, int& iy0, int& iy1,
                                             float& w00, float& w01, float& w10, float& w11) {
  float pxx = (cx + 1.0f) * 0.5f * (RPLANE - 1);
  float pyy = (cy + 1.0f) * 0.5f * (RPLANE - 1);
  float lx = floorf(pxx), ly = floorf(pyy);
  float fx = pxx - lx, fy = pyy - ly;
  int jx = (int)lx, jy = (int)ly;
  ix0 = min(max(jx, 0), RPLANE - 1);
  ix1 = min(max(jx + 1, 0), RPLANE - 1);
  iy0 = min(max(jy, 0), RPLANE - 1);
  iy1 = min(max(jy + 1, 0), RPLANE - 1);
  w00 = (1.0f - fy) * (1.0f - fx);
  w01 = (1.0f - fy) * fx;
  w10 = fy * (1.0f - fx);
  w11 = fy * fx;
}

// ---------------- fused point kernel ----------------------------------------
// 1 wave = one 16-point tile; 8 waves / block; NPTS/128 blocks.
__global__ __launch_bounds__(256) void nerf_point_mlp(
    const float* __restrict__ rays_o, const float* __restrict__ rays_d,
    const float* __restrict__ planes, const float* __restrict__ features,
    const float* __restrict__ bs1, const float* __restrict__ bs2,
    const float* __restrict__ bc1, const float* __restrict__ bc2,
    const _Float16* __restrict__ wfrag,
    float* __restrict__ density_out, float* __restrict__ rgbraw_out) {
  // LDS staged directly in WMMA A-fragment lane layout: [wave][lane][16 elems]
  __shared__ float    sprodLds[8][16][6];
  __shared__ float    maskLds[8][16];
  __shared__ __align__(32) _Float16 featFragLds[8][32][16];
  __shared__ __align__(32) _Float16 hFragLds[8][2][32][16];

  const int lane = threadIdx.x & 31;
  const int wave = threadIdx.x >> 5;
  const int m    = lane & 15;
  const int hi   = lane >> 4;
  const int tile = blockIdx.x * 8 + wave;
  const int p    = tile * 16 + m;
  const int ray  = p >> 8;
  const int samp = p & 255;

  // ---- ray-cube geometry (raw rays_d, as in reference) ----
  const float ox = rays_o[ray * 3 + 0], oy = rays_o[ray * 3 + 1], oz = rays_o[ray * 3 + 2];
  const float dx = rays_d[ray * 3 + 0], dy = rays_d[ray * 3 + 1], dz = rays_d[ray * 3 + 2];
  float dsx = (fabsf(dx) < 1e-6f) ? 1e-6f : dx;
  float dsy = (fabsf(dy) < 1e-6f) ? 1e-6f : dy;
  float dsz = (fabsf(dz) < 1e-6f) ? 1e-6f : dz;
  float t1x = (-kRadius - ox) / dsx, t2x = (kRadius - ox) / dsx;
  float t1y = (-kRadius - oy) / dsy, t2y = (kRadius - oy) / dsy;
  float t1z = (-kRadius - oz) / dsz, t2z = (kRadius - oz) / dsz;
  float tmin = fmaxf(fminf(t1x, t2x), fmaxf(fminf(t1y, t2y), fminf(t1z, t2z)));
  float tmax = fminf(fmaxf(t1x, t2x), fminf(fmaxf(t1y, t2y), fmaxf(t1z, t2z)));
  float nearv = fmaxf(tmin, 0.0f);
  float t = nearv + ((float)samp + 0.5f) * kDT;
  float maskf = ((t < tmax) && (tmax > nearv)) ? 1.0f : 0.0f;
  float px = (ox + t * dx) / kRadius;
  float py = (oy + t * dy) / kRadius;
  float pz = (oz + t * dz) / kRadius;

  // ---- low-rank plane sampling: this lane covers channels c0..c0+2 of 6 ----
  const int c0 = hi * 3;
  float sp0 = 1.0f, sp1 = 1.0f, sp2 = 1.0f;
#define SAMPLE_PLANE(PL, CX, CY)                                                  \
  do {                                                                            \
    int ix0, ix1, iy0, iy1; float w00, w01, w10, w11;                             \
    bilerp_setup((CX), (CY), ix0, ix1, iy0, iy1, w00, w01, w10, w11);             \
    const float* P = planes + ((PL) * 6 + c0) * (RPLANE * RPLANE);                \
    sp0 *= w00 * P[iy0 * RPLANE + ix0] + w01 * P[iy0 * RPLANE + ix1]              \
         + w10 * P[iy1 * RPLANE + ix0] + w11 * P[iy1 * RPLANE + ix1];             \
    P += RPLANE * RPLANE;                                                         \
    sp1 *= w00 * P[iy0 * RPLANE + ix0] + w01 * P[iy0 * RPLANE + ix1]              \
         + w10 * P[iy1 * RPLANE + ix0] + w11 * P[iy1 * RPLANE + ix1];             \
    P += RPLANE * RPLANE;                                                         \
    sp2 *= w00 * P[iy0 * RPLANE + ix0] + w01 * P[iy0 * RPLANE + ix1]              \
         + w10 * P[iy1 * RPLANE + ix0] + w11 * P[iy1 * RPLANE + ix1];             \
  } while (0)
  SAMPLE_PLANE(0, px, py);  // comb (0,1): x -> last grid axis
  SAMPLE_PLANE(1, px, pz);  // comb (0,2)
  SAMPLE_PLANE(2, py, pz);  // comb (1,2)
#undef SAMPLE_PLANE

  sprodLds[wave][m][c0 + 0] = sp0;
  sprodLds[wave][m][c0 + 1] = sp1;
  sprodLds[wave][m][c0 + 2] = sp2;
  maskLds[wave][m] = maskf;  // both halves write identical value

  // learned coords: interp[o] = sum_r prod_p s[p][2o+r]  (same-wave LDS in-order)
  float i0 = sprodLds[wave][m][0] + sprodLds[wave][m][1];
  float i1 = sprodLds[wave][m][2] + sprodLds[wave][m][3];
  float i2 = sprodLds[wave][m][4] + sprodLds[wave][m][5];

  // ---- trilinear sample of feature grid; lane covers 16 of 32 channels ----
  float pxx = (i0 + 1.0f) * 0.5f * (RFEAT - 1);
  float pyy = (i1 + 1.0f) * 0.5f * (RFEAT - 1);
  float pzz = (i2 + 1.0f) * 0.5f * (RFEAT - 1);
  float lx = floorf(pxx), ly = floorf(pyy), lz = floorf(pzz);
  float fx = pxx - lx, fy = pyy - ly, fz = pzz - lz;
  int jx = (int)lx, jy = (int)ly, jz = (int)lz;
  int ix0 = min(max(jx, 0), RFEAT - 1), ix1 = min(max(jx + 1, 0), RFEAT - 1);
  int iy0 = min(max(jy, 0), RFEAT - 1), iy1 = min(max(jy + 1, 0), RFEAT - 1);
  int iz0 = min(max(jz, 0), RFEAT - 1), iz1 = min(max(jz + 1, 0), RFEAT - 1);
  int   off[8];
  float w8[8];
  off[0] = (iz0 * RFEAT + iy0) * RFEAT + ix0; w8[0] = (1 - fz) * (1 - fy) * (1 - fx);
  off[1] = (iz0 * RFEAT + iy0) * RFEAT + ix1; w8[1] = (1 - fz) * (1 - fy) * fx;
  off[2] = (iz0 * RFEAT + iy1) * RFEAT + ix0; w8[2] = (1 - fz) * fy * (1 - fx);
  off[3] = (iz0 * RFEAT + iy1) * RFEAT + ix1; w8[3] = (1 - fz) * fy * fx;
  off[4] = (iz1 * RFEAT + iy0) * RFEAT + ix0; w8[4] = fz * (1 - fy) * (1 - fx);
  off[5] = (iz1 * RFEAT + iy0) * RFEAT + ix1; w8[5] = fz * (1 - fy) * fx;
  off[6] = (iz1 * RFEAT + iy1) * RFEAT + ix0; w8[6] = fz * fy * (1 - fx);
  off[7] = (iz1 * RFEAT + iy1) * RFEAT + ix1; w8[7] = fz * fy * fx;

  // Channel K = hi*16 + c. Fragment position: reader lane (m, c>>3 within this
  // half's K-window), element e = 8*hi + (c&7). Both 8-channel runs contiguous.
  v8h runLo, runHi;
#pragma unroll
  for (int c = 0; c < 16; ++c) {
    int chg = hi * 16 + c;
    const float* F = features + (size_t)chg * (RFEAT * RFEAT * RFEAT);
    float acc = 0.0f;
#pragma unroll
    for (int k = 0; k < 8; ++k) acc += w8[k] * F[off[k]];
    if (c < 8) runLo[c] = (_Float16)acc; else runHi[c - 8] = (_Float16)acc;
  }
  *(v8h*)&featFragLds[wave][m][8 * hi]      = runLo;  // K = hi*16 .. +7
  *(v8h*)&featFragLds[wave][16 + m][8 * hi] = runHi;  // K = hi*16+8 .. +15

  // normalized ray dir for the color branch
  float rn = rsqrtf(dx * dx + dy * dy + dz * dz);
  float rdx = dx * rn, rdy = dy * rn, rdz = dz * rn;

  // =================== WMMA MLP phase (uniform EXEC) ========================
  const int n = m;
  v16h a_feat = *(const v16h*)&featFragLds[wave][lane][0];

  // -- hidden1 = relu(feats @ Ws1 + bs1) --
#pragma unroll
  for (int nt = 0; nt < 4; ++nt) {
    v8f acc = {};
    acc = wmma_f16(a_feat, load_frag(wfrag, nt, lane), acc);
    int col = nt * 16 + n;
    int f = col >> 5, k = col & 31;
    int e = ((k >> 4) << 3) + (k & 7);
    int rl = ((k >> 3) & 1) * 16;   // reader half-wave for this K
    float bias = bs1[col];
#pragma unroll
    for (int v = 0; v < 8; ++v) {
      int row = v + 8 * hi;
      hFragLds[wave][f][rl + row][e] = (_Float16)fmaxf(acc[v] + bias, 0.0f);
    }
  }

  // -- density = relu(h1 @ Ws2 + bs2) * mask --
  v16h a_h_lo = *(const v16h*)&hFragLds[wave][0][lane][0];
  v16h a_h_hi = *(const v16h*)&hFragLds[wave][1][lane][0];
  {
    v8f cd = {};
    cd = wmma_f16(a_h_lo, load_frag(wfrag, 4, lane), cd);
    cd = wmma_f16(a_h_hi, load_frag(wfrag, 5, lane), cd);
    if (n == 0) {
      float b2 = bs2[0];
#pragma unroll
      for (int v = 0; v < 8; ++v) {
        int row = v + 8 * hi;
        density_out[tile * 16 + row] = fmaxf(cd[v] + b2, 0.0f) * maskLds[wave][row];
      }
    }
  }

  // -- hidden2 = relu([feats, rd, 0..] @ Wc1 + bc1) --
  v16h a_rd;   // K-chunk 32..63 of concat: only local K 0..2 nonzero (rd)
#pragma unroll
  for (int e = 0; e < 16; ++e) {
    int k = ((e >> 3) << 4) + (hi << 3) + (e & 7);
    float val = (k == 0) ? rdx : (k == 1) ? rdy : (k == 2) ? rdz : 0.0f;
    a_rd[e] = (_Float16)val;
  }
#pragma unroll
  for (int nt = 0; nt < 4; ++nt) {
    v8f cc = {};
    cc = wmma_f16(a_feat, load_frag(wfrag, 6 + nt, lane), cc);
    cc = wmma_f16(a_rd,   load_frag(wfrag, 10 + nt, lane), cc);
    int col = nt * 16 + n;
    int f = col >> 5, k = col & 31;
    int e = ((k >> 4) << 3) + (k & 7);
    int rl = ((k >> 3) & 1) * 16;
    float bias = bc1[col];
#pragma unroll
    for (int v = 0; v < 8; ++v) {
      int row = v + 8 * hi;
      hFragLds[wave][f][rl + row][e] = (_Float16)fmaxf(cc[v] + bias, 0.0f);  // h1 already consumed
    }
  }

  // -- rgb_raw = h2 @ Wc2 + bc2 --
  v16h a2_lo = *(const v16h*)&hFragLds[wave][0][lane][0];
  v16h a2_hi = *(const v16h*)&hFragLds[wave][1][lane][0];
  {
    v8f cr = {};
    cr = wmma_f16(a2_lo, load_frag(wfrag, 14, lane), cr);
    cr = wmma_f16(a2_hi, load_frag(wfrag, 15, lane), cr);
    if (n < 3) {
      float bias = bc2[n];
#pragma unroll
      for (int v = 0; v < 8; ++v) {
        int row = v + 8 * hi;
        rgbraw_out[(size_t)(tile * 16 + row) * 3 + n] = cr[v] + bias;
      }
    }
  }
}

// ---------------- per-ray compositing ---------------------------------------
__global__ __launch_bounds__(256) void nerf_composite(
    const float* __restrict__ density, const float* __restrict__ rgbraw,
    float* __restrict__ out) {
  int r = blockIdx.x * blockDim.x + threadIdx.x;
  if (r >= NRAYS) return;
  float trans = 1.0f, wsum = 0.0f;
  float rgb0 = 0.0f, rgb1 = 0.0f, rgb2 = 0.0f;
  for (int i = 0; i < NSAMP; ++i) {
    int p = r * NSAMP + i;
    float dens = density[p];                      // mask already applied
    float dist = (i == NSAMP - 1) ? 1e10f : kDT;  // |rays_d_n| == 1
    float alpha = 1.0f - __expf(-dens * dist);
    float w = alpha * trans;
    trans *= (1.0f - alpha + 1e-10f);
    float r0 = rgbraw[(size_t)p * 3 + 0];
    float r1 = rgbraw[(size_t)p * 3 + 1];
    float r2 = rgbraw[(size_t)p * 3 + 2];
    rgb0 += w / (1.0f + __expf(-r0));
    rgb1 += w / (1.0f + __expf(-r1));
    rgb2 += w / (1.0f + __expf(-r2));
    wsum += w;
  }
  float bg = 1.0f - wsum;
  out[r * 3 + 0] = rgb0 + bg;
  out[r * 3 + 1] = rgb1 + bg;
  out[r * 3 + 2] = rgb2 + bg;
}

// ---------------- launch -----------------------------------------------------
extern "C" void kernel_launch(void* const* d_in, const int* in_sizes, int n_in,
                              void* d_out, int out_size, void* d_ws, size_t ws_size,
                              hipStream_t stream) {
  const float* rays_o   = (const float*)d_in[0];
  const float* rays_d   = (const float*)d_in[1];
  const float* planes   = (const float*)d_in[2];
  const float* features = (const float*)d_in[3];
  const float* Ws1 = (const float*)d_in[4];
  const float* bs1 = (const float*)d_in[5];
  const float* Ws2 = (const float*)d_in[6];
  const float* bs2 = (const float*)d_in[7];
  const float* Wc1 = (const float*)d_in[8];
  const float* bc1 = (const float*)d_in[9];
  const float* Wc2 = (const float*)d_in[10];
  const float* bc2 = (const float*)d_in[11];

  // workspace: [ weight fragments 16KB | density 8MB | rgb_raw 24MB ]
  _Float16* wfrag  = (_Float16*)d_ws;                    // NFRAG*32*16 halves
  float* density = (float*)((char*)d_ws + NFRAG * 32 * 16 * sizeof(_Float16));
  float* rgbraw  = density + NPTS;

  prep_weights<<<1, NFRAG * 32, 0, stream>>>(Ws1, Ws2, Wc1, Wc2, wfrag);

  const int blocks = NPTS / 16 / 8;  // 16384 (8 waves / block)
  nerf_point_mlp<<<blocks, 256, 0, stream>>>(
      rays_o, rays_d, planes, features, bs1, bs2, bc1, bc2, wfrag,
      density, rgbraw);

  nerf_composite<<<(NRAYS + 255) / 256, 256, 0, stream>>>(
      density, rgbraw, (float*)d_out);

  (void)in_sizes; (void)n_in; (void)out_size; (void)ws_size;
}
